// SSA_84988812853841
// MI455X (gfx1250) — matrix operand order repaired and checked by reading, not compiled, not run on hardware
//
#include <hip/hip_runtime.h>
#include <hip/hip_bf16.h>

typedef __bf16 bf16;
typedef __attribute__((ext_vector_type(16))) __bf16 v16bf;
typedef __attribute__((ext_vector_type(8)))  __bf16 v8bf;
typedef __attribute__((ext_vector_type(8)))  float   v8f;

#define D_MODEL 1024
#define N_HEADS 16
#define HEADD   64
#define N_BATCH 8
#define SEQLEN  4096
#define N_ROWS  (N_BATCH * SEQLEN)   // 32768

#define WMMA_BF16(A, B, C) \
  __builtin_amdgcn_wmma_f32_16x16x32_bf16(false, (A), false, (B), (short)0, (C), false, false)

// ---- WMMA operand loaders in contiguous k order (k = 16*(lane/16) + s), same for A and B ----

// 16-bit A matrix 16x32 (MxK), row-major source with leading dim `ld`.
// lane<16: M=lane, K = 0..15
// lane>=16: M=lane-16, K = 16..31
static __device__ __forceinline__ v16bf load_A16x32(const bf16* __restrict__ p, int ld, int lane) {
  const int m  = lane & 15;
  const int kb = lane & 16;                  // 0 or 16
  return *(const v16bf*)(p + m * ld + kb);
}

// 16-bit B matrix 32x16 (KxN): B[k][n] = p[n*ld + k]; lane<16: N=lane, K=0..15;
// lane>=16: N=lane-16, K=16..31. One contiguous 32B chunk per lane.
static __device__ __forceinline__ v16bf load_B32x16(const bf16* __restrict__ p, int ld, int lane) {
  const int n  = lane & 15;
  const int kb = lane & 16;                  // 0 or 16
  return *(const v16bf*)(p + n * ld + kb);
}

// f32 C/D 16x16: VGPR r -> row (r or r+8), col = lane&15
static __device__ __forceinline__ void store_D_bf16(bf16* __restrict__ p, int ld, int lane, v8f d) {
  const int n  = lane & 15;
  const int rb = (lane & 16) >> 1;           // 0 or 8
#pragma unroll
  for (int r = 0; r < 8; ++r) p[(rb + r) * ld + n] = (bf16)d[r];
}
static __device__ __forceinline__ void store_D_f32(float* __restrict__ p, int ld, int lane, v8f d) {
  const int n  = lane & 15;
  const int rb = (lane & 16) >> 1;
#pragma unroll
  for (int r = 0; r < 8; ++r) p[(rb + r) * ld + n] = d[r];
}

// ---------------------------------------------------------------------------
// Utility kernels
// ---------------------------------------------------------------------------
__global__ void cast_f32_bf16(const float* __restrict__ src, bf16* __restrict__ dst, int n) {
  int i = blockIdx.x * blockDim.x + threadIdx.x;
  int stride = gridDim.x * blockDim.x;
  for (; i < n; i += stride) dst[i] = (bf16)src[i];
}

__global__ void zero_f32(float* __restrict__ p, int n) {
  int i = blockIdx.x * blockDim.x + threadIdx.x;
  int stride = gridDim.x * blockDim.x;
  for (; i < n; i += stride) p[i] = 0.f;
}

// ---------------------------------------------------------------------------
// C[M,E] = A[M,K] * W[E,K]^T   (M=32768, E=K=1024), bf16 in, f32 accumulate.
// Block: 256 thr = 8 waves; block tile 128(M) x 256(E); wave = 64x64 = 4x4
// accumulators -> 16 WMMAs per 16 global_load_b128 (32 FLOP/loaded byte).
// outb path: write bf16 + fuse per-column sum / sumsq (BatchNorm stats).
// outf path: add bias, write f32 (final projection).
// ---------------------------------------------------------------------------
__global__ __launch_bounds__(256) void gemm_bt_kernel(
    const bf16* __restrict__ A, const bf16* __restrict__ W,
    bf16* __restrict__ outb, float* __restrict__ outf,
    const float* __restrict__ bias,
    float* __restrict__ colsum, float* __restrict__ colsumsq)
{
  const int lane = threadIdx.x & 31;
  const int wave = threadIdx.x >> 5;
  const int e0 = blockIdx.x * 256 + (wave & 3) * 64;   // column base for wave
  const int m0 = blockIdx.y * 128 + (wave >> 2) * 64;  // row base for wave

  v8f acc[4][4];
#pragma unroll
  for (int i = 0; i < 4; ++i)
#pragma unroll
    for (int j = 0; j < 4; ++j) acc[i][j] = {};

  for (int k0 = 0; k0 < D_MODEL; k0 += 32) {
    v16bf a[4], b[4];
#pragma unroll
    for (int i = 0; i < 4; ++i)
      a[i] = load_A16x32(A + (size_t)(m0 + i * 16) * D_MODEL + k0, D_MODEL, lane);
#pragma unroll
    for (int j = 0; j < 4; ++j)
      b[j] = load_B32x16(W + (size_t)(e0 + j * 16) * D_MODEL + k0, D_MODEL, lane);
    if (k0 + 32 < D_MODEL) {  // global_prefetch_b8 for next K slice of A
      __builtin_prefetch(A + (size_t)m0 * D_MODEL + k0 + 32, 0, 1);
    }
#pragma unroll
    for (int i = 0; i < 4; ++i)
#pragma unroll
      for (int j = 0; j < 4; ++j)
        acc[i][j] = WMMA_BF16(a[i], b[j], acc[i][j]);
  }

  if (outb) {
#pragma unroll
    for (int i = 0; i < 4; ++i)
#pragma unroll
      for (int j = 0; j < 4; ++j)
        store_D_bf16(outb + (size_t)(m0 + i * 16) * D_MODEL + (e0 + j * 16),
                     D_MODEL, lane, acc[i][j]);
    // BatchNorm statistics from the f32 accumulators (pre-rounding).
#pragma unroll
    for (int j = 0; j < 4; ++j) {
      float s = 0.f, s2 = 0.f;
#pragma unroll
      for (int i = 0; i < 4; ++i)
#pragma unroll
        for (int r = 0; r < 8; ++r) { float v = acc[i][j][r]; s += v; s2 += v * v; }
      // each lane owns one column (col = lane&15) for half the rows; fold halves
      s  += __shfl_xor(s,  16, 32);
      s2 += __shfl_xor(s2, 16, 32);
      if (lane < 16) {
        atomicAdd(&colsum[e0 + j * 16 + lane], s);
        atomicAdd(&colsumsq[e0 + j * 16 + lane], s2);
      }
    }
  } else {
#pragma unroll
    for (int i = 0; i < 4; ++i)
#pragma unroll
      for (int j = 0; j < 4; ++j) {
        v8f d = acc[i][j];
        const float bv = bias ? bias[e0 + j * 16 + (lane & 15)] : 0.f;
#pragma unroll
        for (int r = 0; r < 8; ++r) d[r] += bv;
        store_D_f32(outf + (size_t)(m0 + i * 16) * D_MODEL + (e0 + j * 16),
                    D_MODEL, lane, d);
      }
  }
}

// ---------------------------------------------------------------------------
// BN finalize: scale = gamma * rsqrt(var+eps), shift = beta - mean*scale
// grid = 3 branches, block = 1024 features
// ---------------------------------------------------------------------------
__global__ void bn_params_kernel(
    const float* __restrict__ sums, const float* __restrict__ sumsq,
    const float* __restrict__ gq, const float* __restrict__ bq,
    const float* __restrict__ gk, const float* __restrict__ bk,
    const float* __restrict__ gv, const float* __restrict__ bv,
    float* __restrict__ scale, float* __restrict__ shift)
{
  const int br = blockIdx.x;
  const int i  = threadIdx.x;
  const float inv = 1.f / (float)N_ROWS;
  float mean = sums[br * 1024 + i] * inv;
  float var  = sumsq[br * 1024 + i] * inv - mean * mean;  // biased, matches BN
  const float* g = (br == 0) ? gq : (br == 1) ? gk : gv;
  const float* b = (br == 0) ? bq : (br == 1) ? bk : bv;
  float sc = g[i] * rsqrtf(var + 1e-5f);
  scale[br * 1024 + i] = sc;
  shift[br * 1024 + i] = b[i] - mean * sc;
}

// ---------------------------------------------------------------------------
// Spike + qk fuse.  One block = 64(n) x 64(feature) tile of one (b,h).
// Writes S = sigmoid(4(hq_n-1))*sigmoid(4(hk_n-1))*hd^-0.5 row-major, and
// spiked V transposed per head to Vt[b,h,j,n] via an LDS staging tile.
// ---------------------------------------------------------------------------
__global__ __launch_bounds__(256) void spike_kernel(
    const bf16* __restrict__ Hq, const bf16* __restrict__ Hk, const bf16* __restrict__ Hv,
    const float* __restrict__ scale, const float* __restrict__ shift,
    bf16* __restrict__ S, bf16* __restrict__ Vt)
{
  __shared__ bf16 tile[64 * 65];
  const int bh = blockIdx.x;            // 0..127  (b*16+h)
  const int n0 = blockIdx.y * 64;       // seq tile
  const int b  = bh >> 4;
  const int h  = bh & 15;
  const int t  = threadIdx.x;

#pragma unroll
  for (int i = 0; i < 16; ++i) {
    const int lin = t + i * 256;
    const int r = lin >> 6, c = lin & 63;
    const int row = b * SEQLEN + n0 + r;
    const int col = h * HEADD + c;
    const size_t idx = (size_t)row * D_MODEL + col;
    float hq = (float)Hq[idx] * scale[col]        + shift[col];
    float hk = (float)Hk[idx] * scale[1024 + col] + shift[1024 + col];
    float hv = (float)Hv[idx] * scale[2048 + col] + shift[2048 + col];
    float q  = 1.f / (1.f + __expf(-4.f * (hq - 1.f)));
    float kk = 1.f / (1.f + __expf(-4.f * (hk - 1.f)));
    float vs = 1.f / (1.f + __expf(-4.f * (hv - 1.f)));
    S[idx] = (bf16)(q * kk * 0.125f);             // * HEADD^-0.5
    tile[c * 65 + r] = (bf16)vs;
  }
  __syncthreads();
#pragma unroll
  for (int i = 0; i < 16; ++i) {
    const int lin = t + i * 256;
    const int c = lin >> 6, r = lin & 63;
    Vt[(size_t)(bh * HEADD + c) * SEQLEN + n0 + r] = tile[c * 65 + r];
  }
}

// ---------------------------------------------------------------------------
// Gram: vv[b,h] = V^T V over N=4096.  One block per (b,h); 8 waves cover the
// 4x4 grid of 16x16 output tiles (2 tiles per wave, shared A operand).
// Both operands come from Vt (contiguous along K=n).
// ---------------------------------------------------------------------------
__global__ __launch_bounds__(256) void gram_kernel(
    const bf16* __restrict__ Vt, bf16* __restrict__ vvb)
{
  const int bh   = blockIdx.x;
  const int lane = threadIdx.x & 31;
  const int wave = threadIdx.x >> 5;
  const int ti  = wave >> 1;            // 0..3  row tile
  const int tj0 = (wave & 1) * 2;       // 0 or 2 : two col tiles
  const bf16* base = Vt + (size_t)bh * HEADD * SEQLEN;

  v8f acc0 = {}, acc1 = {};
  for (int n0 = 0; n0 < SEQLEN; n0 += 32) {
    v16bf a  = load_A16x32(base + (size_t)(ti * 16) * SEQLEN + n0, SEQLEN, lane);
    v16bf b0 = load_B32x16(base + (size_t)(tj0 * 16) * SEQLEN + n0, SEQLEN, lane);
    v16bf b1 = load_B32x16(base + (size_t)((tj0 + 1) * 16) * SEQLEN + n0, SEQLEN, lane);
    acc0 = WMMA_BF16(a, b0, acc0);
    acc1 = WMMA_BF16(a, b1, acc1);
  }
  bf16* o = vvb + (size_t)bh * HEADD * HEADD;
  store_D_bf16(o + (ti * 16) * HEADD + tj0 * 16,       HEADD, lane, acc0);
  store_D_bf16(o + (ti * 16) * HEADD + (tj0 + 1) * 16, HEADD, lane, acc1);
}

// ---------------------------------------------------------------------------
// Apply: Y[b,n,h*64+j] = sum_d S[b,n,h*64+d] * vv[b,h][d,j]
// vv is symmetric, so B[k,n] = vv[n][k] reads vv rows contiguously.
// Block = one (b,h) x 128 rows; 8 waves x (1x4) tiles.
// ---------------------------------------------------------------------------
__global__ __launch_bounds__(256) void apply_vv_kernel(
    const bf16* __restrict__ S, const bf16* __restrict__ vvb, bf16* __restrict__ Y)
{
  const int bh   = blockIdx.x;          // 0..127
  const int mt   = blockIdx.y;          // 0..31 (128-row blocks)
  const int b    = bh >> 4;
  const int h    = bh & 15;
  const int lane = threadIdx.x & 31;
  const int wave = threadIdx.x >> 5;
  const size_t rowbase = (size_t)b * SEQLEN + mt * 128 + wave * 16;
  const bf16* vb = vvb + (size_t)bh * HEADD * HEADD;

  v8f acc[4];
#pragma unroll
  for (int j = 0; j < 4; ++j) acc[j] = {};

#pragma unroll
  for (int k0 = 0; k0 < HEADD; k0 += 32) {
    v16bf a = load_A16x32(S + rowbase * D_MODEL + h * HEADD + k0, D_MODEL, lane);
#pragma unroll
    for (int j = 0; j < 4; ++j) {
      v16bf bt = load_B32x16(vb + (j * 16) * HEADD + k0, HEADD, lane);  // symmetric vv
      acc[j] = WMMA_BF16(a, bt, acc[j]);
    }
  }
#pragma unroll
  for (int j = 0; j < 4; ++j)
    store_D_bf16(Y + rowbase * D_MODEL + h * HEADD + j * 16, D_MODEL, lane, acc[j]);
}

// ---------------------------------------------------------------------------
extern "C" void kernel_launch(void* const* d_in, const int* in_sizes, int n_in,
                              void* d_out, int out_size, void* d_ws, size_t ws_size,
                              hipStream_t stream)
{
  (void)in_sizes; (void)n_in; (void)out_size; (void)ws_size;
  const float* x  = (const float*)d_in[0];
  const float* Wq = (const float*)d_in[1];
  const float* gq = (const float*)d_in[2];
  const float* bq = (const float*)d_in[3];
  const float* Wk = (const float*)d_in[4];
  const float* gk = (const float*)d_in[5];
  const float* bk = (const float*)d_in[6];
  const float* Wv = (const float*)d_in[7];
  const float* gv = (const float*)d_in[8];
  const float* bv = (const float*)d_in[9];
  const float* Wp = (const float*)d_in[10];
  const float* bp = (const float*)d_in[11];
  float* out = (float*)d_out;

  char* ws = (char*)d_ws;
  size_t off = 0;
  auto take = [&](size_t bytes) -> char* {
    char* p = ws + off;
    off += (bytes + 255) & ~(size_t)255;
    return p;
  };

  const size_t big = (size_t)N_ROWS * D_MODEL * sizeof(bf16);  // 64 MiB
  bf16* bufA = (bf16*)take(big);                               // Xb, later S
  bf16* Wqb  = (bf16*)take((size_t)D_MODEL * D_MODEL * 2);
  bf16* Wkb  = (bf16*)take((size_t)D_MODEL * D_MODEL * 2);
  bf16* Wvb  = (bf16*)take((size_t)D_MODEL * D_MODEL * 2);
  bf16* Wpb  = (bf16*)take((size_t)D_MODEL * D_MODEL * 2);
  bf16* Hq   = (bf16*)take(big);                               // later Y
  bf16* Hk   = (bf16*)take(big);
  bf16* Hv   = (bf16*)take(big);
  bf16* Vt   = (bf16*)take(big);                               // [B*H*64, 4096]
  bf16* vvb  = (bf16*)take((size_t)128 * HEADD * HEADD * 2);
  float* sums  = (float*)take(3 * 1024 * sizeof(float));
  float* sumsq = (float*)take(3 * 1024 * sizeof(float));
  float* bnsc  = (float*)take(3 * 1024 * sizeof(float));
  float* bnsh  = (float*)take(3 * 1024 * sizeof(float));

  // 1) casts to bf16
  cast_f32_bf16<<<dim3(4096), dim3(256), 0, stream>>>(x,  bufA, N_ROWS * D_MODEL);
  cast_f32_bf16<<<dim3(512),  dim3(256), 0, stream>>>(Wq, Wqb, D_MODEL * D_MODEL);
  cast_f32_bf16<<<dim3(512),  dim3(256), 0, stream>>>(Wk, Wkb, D_MODEL * D_MODEL);
  cast_f32_bf16<<<dim3(512),  dim3(256), 0, stream>>>(Wv, Wvb, D_MODEL * D_MODEL);
  cast_f32_bf16<<<dim3(512),  dim3(256), 0, stream>>>(Wp, Wpb, D_MODEL * D_MODEL);
  zero_f32<<<dim3(12), dim3(256), 0, stream>>>(sums,  3 * 1024);
  zero_f32<<<dim3(12), dim3(256), 0, stream>>>(sumsq, 3 * 1024);

  // 2) branch GEMMs with fused BN statistics
  dim3 ggrid(D_MODEL / 256, N_ROWS / 128);  // (4, 256)
  gemm_bt_kernel<<<ggrid, 256, 0, stream>>>(bufA, Wqb, Hq, nullptr, nullptr,
                                            sums + 0,    sumsq + 0);
  gemm_bt_kernel<<<ggrid, 256, 0, stream>>>(bufA, Wkb, Hk, nullptr, nullptr,
                                            sums + 1024, sumsq + 1024);
  gemm_bt_kernel<<<ggrid, 256, 0, stream>>>(bufA, Wvb, Hv, nullptr, nullptr,
                                            sums + 2048, sumsq + 2048);

  // 3) BN finalize
  bn_params_kernel<<<dim3(3), dim3(1024), 0, stream>>>(sums, sumsq,
                                                       gq, bq, gk, bk, gv, bv,
                                                       bnsc, bnsh);

  // 4) spike + qk fuse; S reuses the Xb buffer (x no longer needed)
  spike_kernel<<<dim3(128, 64), dim3(256), 0, stream>>>(Hq, Hk, Hv, bnsc, bnsh,
                                                        bufA /*S*/, Vt);

  // 5) per-head Gram matrices
  gram_kernel<<<dim3(128), dim3(256), 0, stream>>>(Vt, vvb);

  // 6) apply vv; Y reuses the Hq buffer
  apply_vv_kernel<<<dim3(128, 32), dim3(256), 0, stream>>>(bufA /*S*/, vvb,
                                                           Hq /*Y*/);

  // 7) output projection + bias, f32 out
  gemm_bt_kernel<<<ggrid, 256, 0, stream>>>(Hq /*Y*/, Wpb, nullptr, out, bp,
                                            nullptr, nullptr);
}
